// SatarkGAT_34806414967336
// MI455X (gfx1250) — compile-verified, matrix operand order checked by reading, not run on hardware
//
#include <hip/hip_runtime.h>
#include <hip/hip_bf16.h>

// ---------------------------------------------------------------------------
// GATv2 (2 layers + classifier) for MI455X (gfx1250, wave32).
// GEMMs use v_wmma_f32_16x16x32_f16; edge phase uses L2-resident gathers and
// float atomics (segment max / sum); softmax fused into 2 edge passes.
// ---------------------------------------------------------------------------

typedef __attribute__((ext_vector_type(16))) _Float16 v16h;
typedef __attribute__((ext_vector_type(8)))  float    v8f;

union V16H { v16h v; uint4 q[2]; };

__device__ __forceinline__ float lrelu(float v) { return v > 0.f ? v : 0.2f * v; }
__device__ __forceinline__ float elu(float v)   { return v > 0.f ? v : (__expf(v) - 1.f); }

// float atomic max via monotonic int encoding (buffer initialized to -inf bits)
__device__ __forceinline__ void atomicMaxFloat(float* addr, float v) {
    if (v >= 0.f) atomicMax((int*)addr, __float_as_int(v));
    else          atomicMin((unsigned int*)addr, __float_as_uint(v));
}

// ---------------------------------------------------------------------------
__global__ void fill_u32_kernel(unsigned int* __restrict__ p, unsigned int v, int n) {
    int t = blockIdx.x * blockDim.x + threadIdx.x;
    if (t < n) p[t] = v;
}

// fp32 -> fp16 elementwise convert
__global__ void conv_f16_kernel(const float* __restrict__ src, _Float16* __restrict__ dst, int n) {
    int t = blockIdx.x * blockDim.x + threadIdx.x;
    if (t < n) dst[t] = (_Float16)src[t];
}

// Pack [Wl | Wr] (each [K x Ch] row-major fp32) into column-major f16 Wt[2*Ch][K],
// and concatenated bias bc[2*Ch].
__global__ void prep_w_kernel(const float* __restrict__ Wl, const float* __restrict__ Wr,
                              const float* __restrict__ bl, const float* __restrict__ br,
                              _Float16* __restrict__ Wt, float* __restrict__ bc,
                              int K, int Ch) {
    int t = blockIdx.x * blockDim.x + threadIdx.x;
    int total = 2 * Ch * K;
    if (t >= total) return;
    int c = t / K, k = t - c * K;
    float w = (c < Ch) ? Wl[k * Ch + c] : Wr[k * Ch + (c - Ch)];
    Wt[c * K + k] = (_Float16)w;
    if (k == 0) bc[c] = (c < Ch) ? bl[c] : br[c - Ch];
}

// ---------------------------------------------------------------------------
// C[M x NCOLS] = A[M x 128](f16, row-major) * Bt[NCOLS x 128](f16, col-major) + bias
// One wave computes one 16x16 tile; K=128 -> 4 x v_wmma_f32_16x16x32_f16.
// blockDim = 128 (4 waves); blockIdx.x -> M tile, blockIdx.y*4+wave -> N tile.
// NCOLS is compile-time so D stores are base + immediate offsets.
template <int NCOLS>
__global__ void __launch_bounds__(128)
gemm_wmma_kernel(const _Float16* __restrict__ A, const _Float16* __restrict__ Bt,
                 const float* __restrict__ bias, float* __restrict__ C, int M) {
    int lane = threadIdx.x & 31;
    int wave = threadIdx.x >> 5;
    int tileN = (blockIdx.y * 4 + wave) * 16;
    int tileM = blockIdx.x * 16;
    if (tileN >= NCOLS) return;              // wave-uniform: EXEC stays all-ones
    int r  = lane & 15;
    int kh = lane >> 4;                      // K-half select per WMMA f16 layout
    int rowA = tileM + r; if (rowA >= M) rowA = M - 1;   // clamp (guard on store)
    const _Float16* ap = A  + (size_t)rowA * 128 + kh * 8;
    const _Float16* bp = Bt + (size_t)(tileN + r) * 128 + kh * 8;
    v8f acc = {};
#pragma unroll
    for (int ks = 0; ks < 128; ks += 32) {
        V16H a, b;
        a.q[0] = *(const uint4*)(ap + ks);
        a.q[1] = *(const uint4*)(ap + ks + 16);
        b.q[0] = *(const uint4*)(bp + ks);
        b.q[1] = *(const uint4*)(bp + ks + 16);
        acc = __builtin_amdgcn_wmma_f32_16x16x32_f16(false, a.v, false, b.v,
                                                     (short)0, acc, false, false);
    }
    int col = tileN + r;
    float bv = bias[col];
    int mbase = tileM + ((lane < 16) ? 0 : 8);
    float* cp = C + (size_t)mbase * NCOLS + col;
    if (tileM + 16 <= M) {                   // wave-uniform full-tile fast path
#pragma unroll
        for (int v = 0; v < 8; ++v) cp[(size_t)v * NCOLS] = acc[v] + bv;
    } else {                                 // ragged last tile
#pragma unroll
        for (int v = 0; v < 8; ++v)
            if (mbase + v < M) cp[(size_t)v * NCOLS] = acc[v] + bv;
    }
}

// ---------------------------------------------------------------------------
// Layer 1 edge pass A: logits[e,h] = att_h . lrelu(xl[src]+xr[dst]); atomic max per (dst,h).
// xlr: [N][256] = [xl | xr]. One wave per edge; lane handles 4 channels (head = lane>>3).
__global__ void edge_logits1_kernel(const float* __restrict__ xlr, const int* __restrict__ ei,
                                    const float* __restrict__ att, float* __restrict__ logits,
                                    float* __restrict__ mbuf, int Eedges, int Etot) {
    int e = blockIdx.x * 8 + (threadIdx.x >> 5);
    if (e >= Etot) return;
    int lane = threadIdx.x & 31;
    int s, d;
    if (e < Eedges) { s = ei[e]; d = ei[Eedges + e]; } else { s = e - Eedges; d = s; }
    const float4 xl = *(const float4*)(xlr + (size_t)s * 256 + lane * 4);
    const float4 xr = *(const float4*)(xlr + (size_t)d * 256 + 128 + lane * 4);
    const float4 av = *(const float4*)(att + lane * 4);
    float t = av.x * lrelu(xl.x + xr.x) + av.y * lrelu(xl.y + xr.y)
            + av.z * lrelu(xl.z + xr.z) + av.w * lrelu(xl.w + xr.w);
    t += __shfl_xor(t, 1); t += __shfl_xor(t, 2); t += __shfl_xor(t, 4);  // reduce 8-lane head
    if ((lane & 7) == 0) {
        int h = lane >> 3;
        logits[(size_t)e * 4 + h] = t;
        atomicMaxFloat(mbuf + (size_t)d * 4 + h, t);
    }
}

// Layer 1 edge pass B: a = exp(logit - m[dst]); denom[dst,h] += a; accum[dst] += xl[src]*a.
__global__ void edge_agg1_kernel(const float* __restrict__ xlr, const int* __restrict__ ei,
                                 const float* __restrict__ logits, const float* __restrict__ mbuf,
                                 float* __restrict__ denom, float* __restrict__ accum,
                                 int Eedges, int Etot) {
    int e = blockIdx.x * 8 + (threadIdx.x >> 5);
    if (e >= Etot) return;
    int lane = threadIdx.x & 31;
    int s, d;
    if (e < Eedges) { s = ei[e]; d = ei[Eedges + e]; } else { s = e - Eedges; d = s; }
    int h = lane >> 3;
    float a = __expf(logits[(size_t)e * 4 + h] - mbuf[(size_t)d * 4 + h]);
    if ((lane & 7) == 0) atomicAdd(denom + (size_t)d * 4 + h, a);
    const float4 xl = *(const float4*)(xlr + (size_t)s * 256 + lane * 4);
    float* acc = accum + (size_t)d * 128 + lane * 4;
    atomicAdd(acc + 0, xl.x * a); atomicAdd(acc + 1, xl.y * a);
    atomicAdd(acc + 2, xl.z * a); atomicAdd(acc + 3, xl.w * a);
}

// Layer 1 node finish: normalize, +bias, ELU, emit f16 for layer-2 GEMM.
__global__ void node_finish1_kernel(const float* __restrict__ accum, const float* __restrict__ denom,
                                    const float* __restrict__ bias, _Float16* __restrict__ h1h, int N) {
    int t = blockIdx.x * blockDim.x + threadIdx.x;
    if (t >= N * 128) return;
    int n = t >> 7, c = t & 127, h = c >> 5;
    float v = accum[t] / (denom[n * 4 + h] + 1e-16f) + bias[c];
    h1h[t] = (_Float16)elu(v);
}

// ---------------------------------------------------------------------------
// Layer 2 (H=1, C=32). xlr2: [N][64] = [xl2 | xr2]. Lane = channel.
__global__ void edge_logits2_kernel(const float* __restrict__ xlr, const int* __restrict__ ei,
                                    const float* __restrict__ att, float* __restrict__ logits,
                                    float* __restrict__ mbuf, int Eedges, int Etot) {
    int e = blockIdx.x * 8 + (threadIdx.x >> 5);
    if (e >= Etot) return;
    int lane = threadIdx.x & 31;
    int s, d;
    if (e < Eedges) { s = ei[e]; d = ei[Eedges + e]; } else { s = e - Eedges; d = s; }
    float xa = xlr[(size_t)s * 64 + lane];
    float xb = xlr[(size_t)d * 64 + 32 + lane];
    float t = att[lane] * lrelu(xa + xb);
    t += __shfl_xor(t, 1); t += __shfl_xor(t, 2); t += __shfl_xor(t, 4);
    t += __shfl_xor(t, 8); t += __shfl_xor(t, 16);
    if (lane == 0) { logits[e] = t; atomicMaxFloat(mbuf + d, t); }
}

__global__ void edge_agg2_kernel(const float* __restrict__ xlr, const int* __restrict__ ei,
                                 const float* __restrict__ logits, const float* __restrict__ mbuf,
                                 float* __restrict__ denom, float* __restrict__ accum,
                                 int Eedges, int Etot) {
    int e = blockIdx.x * 8 + (threadIdx.x >> 5);
    if (e >= Etot) return;
    int lane = threadIdx.x & 31;
    int s, d;
    if (e < Eedges) { s = ei[e]; d = ei[Eedges + e]; } else { s = e - Eedges; d = s; }
    float a = __expf(logits[e] - mbuf[d]);
    if (lane == 0) atomicAdd(denom + d, a);
    atomicAdd(accum + (size_t)d * 32 + lane, xlr[(size_t)s * 64 + lane] * a);
}

// Layer 2 node finish + classifier: ELU then sigmoid(h2 . Wc + bc). One wave / node.
__global__ void node_out_kernel(const float* __restrict__ accum, const float* __restrict__ denom,
                                const float* __restrict__ bias, const float* __restrict__ Wc,
                                const float* __restrict__ bc, float* __restrict__ out, int N) {
    int n = blockIdx.x * 8 + (threadIdx.x >> 5);
    if (n >= N) return;
    int lane = threadIdx.x & 31;
    float v = accum[(size_t)n * 32 + lane] / (denom[n] + 1e-16f) + bias[lane];
    float t = elu(v) * Wc[lane];
    t += __shfl_xor(t, 1); t += __shfl_xor(t, 2); t += __shfl_xor(t, 4);
    t += __shfl_xor(t, 8); t += __shfl_xor(t, 16);
    if (lane == 0) out[n] = 1.f / (1.f + __expf(-(t + bc[0])));
}

// ---------------------------------------------------------------------------
extern "C" void kernel_launch(void* const* d_in, const int* in_sizes, int n_in,
                              void* d_out, int out_size, void* d_ws, size_t ws_size,
                              hipStream_t stream) {
    const float* x     = (const float*)d_in[0];
    const int*   ei    = (const int*)  d_in[1];
    const float* W1l   = (const float*)d_in[2];
    const float* b1l   = (const float*)d_in[3];
    const float* W1r   = (const float*)d_in[4];
    const float* b1r   = (const float*)d_in[5];
    const float* att1  = (const float*)d_in[6];
    const float* bias1 = (const float*)d_in[7];
    const float* W2l   = (const float*)d_in[8];
    const float* b2l   = (const float*)d_in[9];
    const float* W2r   = (const float*)d_in[10];
    const float* b2r   = (const float*)d_in[11];
    const float* att2  = (const float*)d_in[12];
    const float* bias2 = (const float*)d_in[13];
    const float* Wc    = (const float*)d_in[14];
    const float* bcv   = (const float*)d_in[15];
    float* out = (float*)d_out;

    const int N      = in_sizes[0] / 128;
    const int Eedges = in_sizes[1] / 2;
    const int Etot   = Eedges + N;

    // ---- workspace carve-up (256B aligned) ----
    char* base = (char*)d_ws;
    size_t off = 0;
    auto algn = [](size_t b) { return (b + 255) & ~(size_t)255; };
    auto take = [&](size_t bytes) -> void* { void* p = base + off; off += algn(bytes); return p; };

    _Float16* xh   = (_Float16*)take((size_t)N * 128 * 2);   // x f16; reused as h1 f16
    _Float16* Wt1  = (_Float16*)take((size_t)256 * 128 * 2);
    float*    bc1  = (float*)   take(256 * 4);
    _Float16* Wt2  = (_Float16*)take((size_t)64 * 128 * 2);
    float*    bc2  = (float*)   take(64 * 4);
    float*    xlr1 = (float*)   take((size_t)N * 256 * 4);   // [xl1 | xr1]
    float*    m1   = (float*)   take((size_t)N * 4 * 4);
    float*    den1 = (float*)   take((size_t)N * 4 * 4);
    float*    acc1 = (float*)   take((size_t)N * 128 * 4);

    // union block: layer-1 logits reused for all layer-2 buffers (ordering-safe)
    size_t s_xlr2 = algn((size_t)N * 64 * 4);
    size_t s_lg2  = algn((size_t)Etot * 4);
    size_t s_m2   = algn((size_t)N * 4);
    size_t s_d2   = algn((size_t)N * 4);
    size_t s_a2   = algn((size_t)N * 32 * 4);
    size_t l2tot  = s_xlr2 + s_lg2 + s_m2 + s_d2 + s_a2;
    size_t l1lg   = algn((size_t)Etot * 4 * 4);
    char* ub = (char*)take(l2tot > l1lg ? l2tot : l1lg);
    float* lg1  = (float*)ub;
    float* xlr2 = (float*)ub;
    float* lg2  = (float*)(ub + s_xlr2);
    float* m2   = (float*)(ub + s_xlr2 + s_lg2);
    float* den2 = (float*)(ub + s_xlr2 + s_lg2 + s_m2);
    float* acc2 = (float*)(ub + s_xlr2 + s_lg2 + s_m2 + s_d2);

    const unsigned NEGINF = 0xFF800000u;
    const int B = 256;
    auto nb = [](int n, int b) { return (n + b - 1) / b; };

    // ---- prep: weights -> col-major f16, x -> f16; init layer-1 accumulators ----
    prep_w_kernel<<<nb(256 * 128, B), B, 0, stream>>>(W1l, W1r, b1l, b1r, Wt1, bc1, 128, 128);
    prep_w_kernel<<<nb(64 * 128, B), B, 0, stream>>>(W2l, W2r, b2l, b2r, Wt2, bc2, 128, 32);
    conv_f16_kernel<<<nb(N * 128, B), B, 0, stream>>>(x, xh, N * 128);
    fill_u32_kernel<<<nb(N * 4, B), B, 0, stream>>>((unsigned*)m1, NEGINF, N * 4);
    fill_u32_kernel<<<nb(N * 4, B), B, 0, stream>>>((unsigned*)den1, 0u, N * 4);
    fill_u32_kernel<<<nb(N * 128, B), B, 0, stream>>>((unsigned*)acc1, 0u, N * 128);

    // ---- layer 1 ----
    int mt = nb(N, 16);
    gemm_wmma_kernel<256><<<dim3(mt, 4), 128, 0, stream>>>(xh, Wt1, bc1, xlr1, N);
    edge_logits1_kernel<<<nb(Etot, 8), B, 0, stream>>>(xlr1, ei, att1, lg1, m1, Eedges, Etot);
    edge_agg1_kernel<<<nb(Etot, 8), B, 0, stream>>>(xlr1, ei, lg1, m1, den1, acc1, Eedges, Etot);
    node_finish1_kernel<<<nb(N * 128, B), B, 0, stream>>>(acc1, den1, bias1, xh, N);

    // ---- layer 2 (init after last read of lg1) ----
    fill_u32_kernel<<<nb(N, B), B, 0, stream>>>((unsigned*)m2, NEGINF, N);
    fill_u32_kernel<<<nb(N, B), B, 0, stream>>>((unsigned*)den2, 0u, N);
    fill_u32_kernel<<<nb(N * 32, B), B, 0, stream>>>((unsigned*)acc2, 0u, N * 32);
    gemm_wmma_kernel<64><<<dim3(mt, 1), 128, 0, stream>>>(xh, Wt2, bc2, xlr2, N);
    edge_logits2_kernel<<<nb(Etot, 8), B, 0, stream>>>(xlr2, ei, att2, lg2, m2, Eedges, Etot);
    edge_agg2_kernel<<<nb(Etot, 8), B, 0, stream>>>(xlr2, ei, lg2, m2, den2, acc2, Eedges, Etot);
    node_out_kernel<<<nb(N, 8), B, 0, stream>>>(acc2, den2, bias2, Wc, bcv, out, N);
}
